// _GEN_NETWORK_20461224198395
// MI455X (gfx1250) — matrix operand order repaired and checked by reading, not compiled, not run on hardware
//
#include <hip/hip_runtime.h>

// ---------------------------------------------------------------------------
// Problem constants (from reference)
// ---------------------------------------------------------------------------
constexpr int B_  = 64;
constexpr int T_  = 64;
constexpr int E_  = 256;
constexpr int H_  = 512;
constexpr int L_  = 256;
constexpr int V_  = 32000;
constexpr int G3_ = 3 * H_;   // 1536
constexpr int NBLK = 8;       // cooperative blocks in scan kernel

typedef __bf16 bf16_t;
typedef __attribute__((ext_vector_type(16))) __bf16 v16bf;
typedef __attribute__((ext_vector_type(8)))  __bf16 v8bf;
typedef __attribute__((ext_vector_type(8)))  float  v8f;
typedef __attribute__((ext_vector_type(4)))  int    v4i_t;

union FragBF { v16bf v; v8bf h[2]; };

__device__ __forceinline__ v8f wmma_bf16(v16bf a, v16bf b, v8f c) {
  // D = A(16x32 bf16) * B(32x16 bf16) + C(16x16 f32)
  return __builtin_amdgcn_wmma_f32_16x16x32_bf16(
      /*neg_a=*/false, a, /*neg_b=*/false, b,
      /*c_mod=*/(short)0, c, /*reuse_a=*/false, /*reuse_b=*/false);
}

__device__ __forceinline__ float sigmoidf_(float x) {
  return 1.0f / (1.0f + __expf(-x));
}

// ---------------------------------------------------------------------------
// CDNA5 async global->LDS copy (ASYNCcnt-tracked DMA), with sync fallback.
// Probe round 2: builtin exists; param 0 is int4 addrspace(1)*, so pass
// explicitly address-space-qualified int4 pointers.
// ---------------------------------------------------------------------------
#if defined(__has_builtin)
#if __has_builtin(__builtin_amdgcn_global_load_async_to_lds_b128)
#define HAVE_ASYNC_LDS 1
#endif
#endif
#ifndef HAVE_ASYNC_LDS
#define HAVE_ASYNC_LDS 0
#endif

__device__ __forceinline__ void async_copy16(const bf16_t* __restrict__ g,
                                             bf16_t* __restrict__ l) {
#if HAVE_ASYNC_LDS
  __builtin_amdgcn_global_load_async_to_lds_b128(
      (__attribute__((address_space(1))) v4i_t*)g,
      (__attribute__((address_space(3))) v4i_t*)l,
      /*offset=*/0, /*cpol=*/0);
#else
  *(v8bf*)l = *(const v8bf*)g;
#endif
}

__device__ __forceinline__ void wait_async0() {
#if HAVE_ASYNC_LDS
#if __has_builtin(__builtin_amdgcn_s_wait_asynccnt)
  __builtin_amdgcn_s_wait_asynccnt(0);
#else
  asm volatile("s_wait_asynccnt 0" ::: "memory");
#endif
#endif
}

// ---------------------------------------------------------------------------
// Grid-wide sync for the cooperative scan kernel (monotonic counter)
// ---------------------------------------------------------------------------
__device__ __forceinline__ void grid_sync(unsigned* cnt, unsigned target) {
  __syncthreads();
  if (threadIdx.x == 0) {
    __builtin_amdgcn_fence(__ATOMIC_RELEASE, "agent");
    __hip_atomic_fetch_add(cnt, 1u, __ATOMIC_RELAXED, __HIP_MEMORY_SCOPE_AGENT);
    while (__hip_atomic_load(cnt, __ATOMIC_RELAXED, __HIP_MEMORY_SCOPE_AGENT) < target) {
      __builtin_amdgcn_s_sleep(1);
    }
  }
  __syncthreads();
  __builtin_amdgcn_fence(__ATOMIC_ACQUIRE, "agent");
}

__global__ void init_kernel(unsigned* cnt) { *cnt = 0u; }

// ---------------------------------------------------------------------------
// f32 -> bf16 strided conversion (weights)
// ---------------------------------------------------------------------------
__global__ void convert_bf16_kernel(const float* __restrict__ src,
                                    bf16_t* __restrict__ dst,
                                    int rows, int cols, int src_ld) {
  int i = blockIdx.x * blockDim.x + threadIdx.x;
  int n = rows * cols;
  if (i < n) {
    int r = i / cols, c = i - r * cols;
    dst[i] = (bf16_t)src[(long)r * src_ld + c];
  }
}

// Gather token embeddings -> bf16 [B*T, E]
__global__ void gather_emb_kernel(const int* __restrict__ seq,
                                  const float* __restrict__ emb,
                                  bf16_t* __restrict__ out) {
  int i = blockIdx.x;      // 0..B*T-1  (i = b*T + t)
  int e = threadIdx.x;     // 0..E-1
  out[i * E_ + e] = (bf16_t)emb[(long)seq[i] * E_ + e];
}

// ---------------------------------------------------------------------------
// Per-batch hoisted small GEMMs (tiny, VALU is fine)
// pu = uh@Wl.T, pi = ih@Wl.T, cu = uh@Wa1[:,H:].T+ba1, ci = ih@Wa1[:,H:].T+ba1
// ---------------------------------------------------------------------------
__global__ void prep1_kernel(const int* __restrict__ uid, const int* __restrict__ iid,
                             const float* __restrict__ uemb, const float* __restrict__ iemb,
                             const float* __restrict__ Wl, const float* __restrict__ Wa1,
                             const float* __restrict__ ba1,
                             float* __restrict__ pu, float* __restrict__ pi,
                             float* __restrict__ cu, float* __restrict__ ci) {
  int which = blockIdx.y;  // 0=pu 1=pi 2=cu 3=ci
  int b = blockIdx.x;
  const float* vec = (which == 0 || which == 2) ? &uemb[(long)uid[b] * L_]
                                                : &iemb[(long)iid[b] * L_];
  if (which < 2) {
    for (int e = threadIdx.x; e < E_; e += blockDim.x) {
      float s = 0.f;
      const float* w = &Wl[(long)e * L_];
      for (int l = 0; l < L_; ++l) s += vec[l] * w[l];
      (which == 0 ? pu : pi)[b * E_ + e] = s;
    }
  } else {
    for (int n = threadIdx.x; n < H_; n += blockDim.x) {
      float s = ba1[n];
      const float* w = &Wa1[(long)n * (H_ + L_) + H_];
      for (int l = 0; l < L_; ++l) s += vec[l] * w[l];
      (which == 2 ? cu : ci)[b * H_ + n] = s;
    }
  }
}

// gu = pu@Wih.T, gp = pi@Wih.T, gbb = bl@Wih.T + bih
__global__ void prep2_kernel(const float* __restrict__ pu, const float* __restrict__ pi,
                             const float* __restrict__ bl, const float* __restrict__ Wih,
                             const float* __restrict__ bih,
                             float* __restrict__ gu, float* __restrict__ gp,
                             float* __restrict__ gbb) {
  int n = blockIdx.x * blockDim.x + threadIdx.x;
  if (n >= G3_) return;
  const float* w = &Wih[(long)n * E_];
  float sb = bih[n];
  for (int e = 0; e < E_; ++e) sb += bl[e] * w[e];
  gbb[n] = sb;
  for (int b = 0; b < B_; ++b) {
    const float* pr = &pu[b * E_];
    const float* qr = &pi[b * E_];
    float su = 0.f, sp = 0.f;
    for (int e = 0; e < E_; ++e) { su += pr[e] * w[e]; sp += qr[e] * w[e]; }
    gu[b * G3_ + n] = su;
    gp[b * G3_ + n] = sp;
  }
}

// ---------------------------------------------------------------------------
// bf16 WMMA GEMM:  C[M,N](f32) = A[M,K](bf16,row) @ B[N,K](bf16,row)^T + bias
// Block tile 64x128, K-chunks of 64, DOUBLE-BUFFERED LDS filled by
// global_load_async_to_lds_b128 (overlaps DMA with WMMA). 8 waves of 16x64.
// Nontemporal C stores: the 524 MB logits stream must not evict Wv from L2.
// M multiple of 64, N multiple of 128, K multiple of 64 (all guaranteed).
// ---------------------------------------------------------------------------
constexpr int BMT = 64, BNT = 128, BKT = 64, LDSK = BKT + 8;

__global__ __launch_bounds__(256)
void gemm_bf16_kernel(const bf16_t* __restrict__ A, int lda,
                      const bf16_t* __restrict__ Bm, int ldb,
                      float* __restrict__ C, int ldc,
                      const float* __restrict__ bias, int K) {
  __shared__ __align__(16) bf16_t Al[2][BMT * LDSK];   // 2 x  9 KB
  __shared__ __align__(16) bf16_t Bl[2][BNT * LDSK];   // 2 x 18 KB
  const int tid  = threadIdx.x;
  const int lane = tid & 31;
  const int wv   = tid >> 5;   // 0..7
  const int wm   = wv >> 1;    // 0..3 -> 16-row slab
  const int wn   = wv & 1;     // 0..1 -> 64-col slab
  const long rowA0 = (long)blockIdx.y * BMT;
  const long colB0 = (long)blockIdx.x * BNT;

  // Issue async DMA for one K-chunk into buffer `buf`.
  auto prefetch = [&](int kk, int buf) {
#pragma unroll
    for (int it = 0; it < 2; ++it) {          // A: 64x64 bf16 = 512 x 16B
      int idx = tid + it * 256;
      int r = idx >> 3, c = (idx & 7) * 8;
      async_copy16(&A[(rowA0 + r) * lda + kk + c], &Al[buf][r * LDSK + c]);
    }
#pragma unroll
    for (int it = 0; it < 4; ++it) {          // B: 128x64 bf16 = 1024 x 16B
      int idx = tid + it * 256;
      int r = idx >> 3, c = (idx & 7) * 8;
      async_copy16(&Bm[(colB0 + r) * ldb + kk + c], &Bl[buf][r * LDSK + c]);
    }
  };

  v8f acc[4];
  v8f vzero = {};
#pragma unroll
  for (int nj = 0; nj < 4; ++nj) acc[nj] = vzero;

  int cur = 0;
  prefetch(0, cur);

  for (int kk = 0; kk < K; kk += BKT) {
    wait_async0();
    __syncthreads();                       // buf[cur] fully in LDS, all waves
    if (kk + BKT < K) prefetch(kk + BKT, cur ^ 1);   // DMA next chunk now

    const bf16_t* Ac = &Al[cur][0];
    const bf16_t* Bc = &Bl[cur][0];
#pragma unroll
    for (int ks = 0; ks < BKT; ks += 32) {
      const int kh = (lane >> 4) * 8;      // A half-wave K offset
      const int kb = (lane >> 4) * 8;      // B half-wave K offset
      FragBF a, b[4];
      {
        int r = wm * 16 + (lane & 15);
        a.h[0] = *(const v8bf*)&Ac[r * LDSK + ks + kh];
        a.h[1] = *(const v8bf*)&Ac[r * LDSK + ks + kh + 16];
      }
#pragma unroll
      for (int nj = 0; nj < 4; ++nj) {
        int cb = wn * 64 + nj * 16 + (lane & 15);
        b[nj].h[0] = *(const v8bf*)&Bc[cb * LDSK + ks + kb];
        b[nj].h[1] = *(const v8bf*)&Bc[cb * LDSK + ks + kb + 16];
      }
#pragma unroll
      for (int nj = 0; nj < 4; ++nj)
        acc[nj] = wmma_bf16(a.v, b[nj].v, acc[nj]);
    }
    __syncthreads();                       // done reading buf[cur]
    cur ^= 1;
  }

#pragma unroll
  for (int nj = 0; nj < 4; ++nj) {
    long col = colB0 + wn * 64 + nj * 16 + (lane & 15);
    float bvv = bias ? bias[col] : 0.f;
#pragma unroll
    for (int vi = 0; vi < 8; ++vi) {
      long row = rowA0 + wm * 16 + (lane >> 4) * 8 + vi;
      __builtin_nontemporal_store(acc[nj][vi] + bvv, &C[row * ldc + col]);
    }
  }
}

// ---------------------------------------------------------------------------
// Cooperative GRU+attention scan. 8 blocks x 384 threads (12 waves).
// Block w owns hidden columns [w*64, w*64+64).
// Per step: Stage A gh=h@Whh.T (WMMA) -> Stage B gating -> grid sync ->
//           Stage C t1=h@Wa1h.T (WMMA) + partial attn dots -> grid sync ->
//           Stage D softmax coefficients (redundant per block).
// ---------------------------------------------------------------------------
__global__ __launch_bounds__(384)
void scan_kernel(const float* __restrict__ Xall,    // [B*T, 3H]
                 const float* __restrict__ gu,      // [B, 3H]
                 const float* __restrict__ gp,      // [B, 3H]
                 const float* __restrict__ gbb,     // [3H]
                 const float* __restrict__ bhh,     // [3H]
                 const float* __restrict__ cu,      // [B, H]
                 const float* __restrict__ ci,      // [B, H]
                 const float* __restrict__ Wa2,     // [H]
                 const float* __restrict__ ba2,     // [1]
                 const bf16_t* __restrict__ Whh_bf, // [3H, H]
                 const bf16_t* __restrict__ Wa1h_bf,// [H, H]
                 bf16_t* __restrict__ Hbf,          // [B*T, H]  (output + carry)
                 float* __restrict__ part_su,       // [NBLK*64]
                 float* __restrict__ part_si,       // [NBLK*64]
                 unsigned* __restrict__ cnt) {
  __shared__ __align__(16) float smem[3 * 64 * 64]; // gh slice; reused for t1
  __shared__ float coef_u[64], coef_i[64];
  const int tid  = threadIdx.x;
  const int lane = tid & 31;
  const int wvi  = tid >> 5;    // 0..11
  const int w    = blockIdx.x;  // 0..7
  unsigned sync_no = 0;
  v8f vzero = {};

  if (tid < 64) { coef_u[tid] = 1.f; coef_i[tid] = 1.f; }
  __syncthreads();

  for (int t = 0; t < T_; ++t) {
    // ---------------- Stage A: gh = h_{t-1} @ Whh.T (3 gate tiles) ---------
    if (t > 0) {
      const int g = wvi >> 2;   // gate 0..2
      const int m = wvi & 3;    // 16-row batch slab 0..3
      v8f acc[4];
#pragma unroll
      for (int nj = 0; nj < 4; ++nj) acc[nj] = vzero;
      const int rA = m * 16 + (lane & 15);
      const long abase = ((long)rA * T_ + (t - 1)) * H_;
      const int kh = (lane >> 4) * 8;
      const int kb = (lane >> 4) * 8;
      for (int ks = 0; ks < H_; ks += 32) {
        FragBF a;
        a.h[0] = *(const v8bf*)&Hbf[abase + ks + kh];
        a.h[1] = *(const v8bf*)&Hbf[abase + ks + kh + 16];
#pragma unroll
        for (int nj = 0; nj < 4; ++nj) {
          int n = g * H_ + w * 64 + nj * 16 + (lane & 15);
          FragBF b;
          b.h[0] = *(const v8bf*)&Whh_bf[(long)n * H_ + ks + kb];
          b.h[1] = *(const v8bf*)&Whh_bf[(long)n * H_ + ks + kb + 16];
          acc[nj] = wmma_bf16(a.v, b.v, acc[nj]);
        }
      }
#pragma unroll
      for (int nj = 0; nj < 4; ++nj)
#pragma unroll
        for (int vi = 0; vi < 8; ++vi) {
          int row = m * 16 + (lane >> 4) * 8 + vi;
          int col = nj * 16 + (lane & 15);
          smem[(g * 64 + row) * 64 + col] = acc[nj][vi];
        }
    } else {
      for (int i = tid; i < 3 * 64 * 64; i += 384) smem[i] = 0.f;
    }
    __syncthreads();

    // ---------------- Stage B: gating -> h_t --------------------------------
    for (int idx = tid; idx < 64 * 64; idx += 384) {
      int b = idx >> 6;
      int c = idx & 63;
      int colH = w * 64 + c;
      float a_u = coef_u[b], a_i = coef_i[b];
      float gi[3], gh[3];
#pragma unroll
      for (int g = 0; g < 3; ++g) {
        int n = g * H_ + colH;
        gi[g] = Xall[((long)b * T_ + t) * G3_ + n]
              + a_u * gu[b * G3_ + n] + a_i * gp[b * G3_ + n] + gbb[n];
        gh[g] = smem[(g * 64 + b) * 64 + c] + bhh[n];
      }
      float r  = sigmoidf_(gi[0] + gh[0]);
      float z  = sigmoidf_(gi[1] + gh[1]);
      float nn = tanhf(gi[2] + r * gh[2]);
      float hp = (t == 0) ? 0.f : (float)Hbf[((long)b * T_ + (t - 1)) * H_ + colH];
      float hv = (1.f - z) * nn + z * hp;
      Hbf[((long)b * T_ + t) * H_ + colH] = (bf16_t)hv;
    }
    grid_sync(cnt, (++sync_no) * NBLK);

    // ---------------- Stage C: t1 = h_t @ Wa1h.T + partial attn dots --------
    if (wvi < 4) {
      const int m = wvi;
      v8f acc[4];
#pragma unroll
      for (int nj = 0; nj < 4; ++nj) acc[nj] = vzero;
      const int rA = m * 16 + (lane & 15);
      const long abase = ((long)rA * T_ + t) * H_;
      const int kh = (lane >> 4) * 8;
      const int kb = (lane >> 4) * 8;
      for (int ks = 0; ks < H_; ks += 32) {
        FragBF a;
        a.h[0] = *(const v8bf*)&Hbf[abase + ks + kh];
        a.h[1] = *(const v8bf*)&Hbf[abase + ks + kh + 16];
#pragma unroll
        for (int nj = 0; nj < 4; ++nj) {
          int n = w * 64 + nj * 16 + (lane & 15);
          FragBF b;
          b.h[0] = *(const v8bf*)&Wa1h_bf[(long)n * H_ + ks + kb];
          b.h[1] = *(const v8bf*)&Wa1h_bf[(long)n * H_ + ks + kb + 16];
          acc[nj] = wmma_bf16(a.v, b.v, acc[nj]);
        }
      }
#pragma unroll
      for (int nj = 0; nj < 4; ++nj)
#pragma unroll
        for (int vi = 0; vi < 8; ++vi) {
          int row = m * 16 + (lane >> 4) * 8 + vi;
          int col = nj * 16 + (lane & 15);
          smem[row * 64 + col] = acc[nj][vi];
        }
    }
    __syncthreads();

    if (tid < 64) {
      int b = tid;
      float su = 0.f, si = 0.f;
      for (int c = 0; c < 64; ++c) {
        int colg = w * 64 + c;
        float t1 = smem[b * 64 + c];
        float wa = Wa2[colg];
        su += tanhf(t1 + cu[b * H_ + colg]) * wa;
        si += tanhf(t1 + ci[b * H_ + colg]) * wa;
      }
      part_su[w * 64 + b] = su;
      part_si[w * 64 + b] = si;
    }
    grid_sync(cnt, (++sync_no) * NBLK);

    // ---------------- Stage D: softmax coefficients (redundant) -------------
    if (tid < 64) {
      int b = tid;
      float su = ba2[0], si = ba2[0];
      for (int ww = 0; ww < NBLK; ++ww) {
        su += part_su[ww * 64 + b];
        si += part_si[ww * 64 + b];
      }
      float mx = fmaxf(su, si);
      float eu = __expf(su - mx), ei = __expf(si - mx);
      float inv = 1.f / (eu + ei);
      coef_u[b] = eu * inv;
      coef_i[b] = ei * inv;
    }
    __syncthreads();
  }
}

// ---------------------------------------------------------------------------
// Host-side orchestration
// ---------------------------------------------------------------------------
extern "C" void kernel_launch(void* const* d_in, const int* in_sizes, int n_in,
                              void* d_out, int out_size, void* d_ws, size_t ws_size,
                              hipStream_t stream) {
  (void)in_sizes; (void)n_in; (void)out_size; (void)ws_size;
  const int*   seq  = (const int*)  d_in[0];
  const int*   uid  = (const int*)  d_in[1];
  const int*   iid  = (const int*)  d_in[2];
  // d_in[3] = random_flag (unused by reference math)
  const float* emb  = (const float*)d_in[4];
  const float* uemb = (const float*)d_in[5];
  const float* iemb = (const float*)d_in[6];
  const float* Wl   = (const float*)d_in[7];
  const float* bl   = (const float*)d_in[8];
  const float* Wih  = (const float*)d_in[9];
  const float* Whh  = (const float*)d_in[10];
  const float* bih  = (const float*)d_in[11];
  const float* bhh  = (const float*)d_in[12];
  const float* Wa1  = (const float*)d_in[13];
  const float* ba1  = (const float*)d_in[14];
  const float* Wa2  = (const float*)d_in[15];
  const float* ba2  = (const float*)d_in[16];
  const float* Wv   = (const float*)d_in[17];
  const float* bv   = (const float*)d_in[18];
  float* out = (float*)d_out;

  // Workspace carve-up (256B-aligned slabs)
  char* base = (char*)d_ws;
  size_t off = 0;
  auto alloc = [&](size_t bytes) -> void* {
    void* p = base + off;
    off += (bytes + 255) & ~(size_t)255;
    return p;
  };
  unsigned* cnt    = (unsigned*)alloc(sizeof(unsigned));
  float* part_su   = (float*)alloc(NBLK * 64 * sizeof(float));
  float* part_si   = (float*)alloc(NBLK * 64 * sizeof(float));
  float* pu        = (float*)alloc((size_t)B_ * E_ * sizeof(float));
  float* pi        = (float*)alloc((size_t)B_ * E_ * sizeof(float));
  float* cu        = (float*)alloc((size_t)B_ * H_ * sizeof(float));
  float* ci        = (float*)alloc((size_t)B_ * H_ * sizeof(float));
  float* gu        = (float*)alloc((size_t)B_ * G3_ * sizeof(float));
  float* gp        = (float*)alloc((size_t)B_ * G3_ * sizeof(float));
  float* gbb       = (float*)alloc((size_t)G3_ * sizeof(float));
  float* Xall      = (float*)alloc((size_t)B_ * T_ * G3_ * sizeof(float));      // 24 MB
  bf16_t* Xemb_bf  = (bf16_t*)alloc((size_t)B_ * T_ * E_ * sizeof(bf16_t));     // 2 MB
  bf16_t* Wih_bf   = (bf16_t*)alloc((size_t)G3_ * E_ * sizeof(bf16_t));
  bf16_t* Whh_bf   = (bf16_t*)alloc((size_t)G3_ * H_ * sizeof(bf16_t));
  bf16_t* Wa1h_bf  = (bf16_t*)alloc((size_t)H_ * H_ * sizeof(bf16_t));
  bf16_t* Wv_bf    = (bf16_t*)alloc((size_t)V_ * H_ * sizeof(bf16_t));          // 32 MB
  bf16_t* Hbf      = (bf16_t*)alloc((size_t)B_ * T_ * H_ * sizeof(bf16_t));     // 4 MB

  // 0) reset cooperative sync counter
  init_kernel<<<1, 1, 0, stream>>>(cnt);

  // 1) weight conversions to bf16
  auto cvt = [&](const float* s, bf16_t* d, int rows, int cols, int ld) {
    int n = rows * cols;
    convert_bf16_kernel<<<(n + 255) / 256, 256, 0, stream>>>(s, d, rows, cols, ld);
  };
  cvt(Wih, Wih_bf, G3_, E_, E_);
  cvt(Whh, Whh_bf, G3_, H_, H_);
  cvt(Wa1, Wa1h_bf, H_, H_, H_ + L_);   // Wa1[:, :H]
  cvt(Wv,  Wv_bf,  V_, H_, H_);

  // 2) gather token embeddings (bf16)
  gather_emb_kernel<<<B_ * T_, E_, 0, stream>>>(seq, emb, Xemb_bf);

  // 3) per-batch hoisted terms
  prep1_kernel<<<dim3(B_, 4), 256, 0, stream>>>(uid, iid, uemb, iemb, Wl, Wa1, ba1,
                                                pu, pi, cu, ci);
  prep2_kernel<<<(G3_ + 255) / 256, 256, 0, stream>>>(pu, pi, bl, Wih, bih, gu, gp, gbb);

  // 4) Xall = x_emb @ Wih.T  (bf16 WMMA GEMM, M=4096 N=1536 K=256)
  gemm_bf16_kernel<<<dim3(G3_ / BNT, (B_ * T_) / BMT), 256, 0, stream>>>(
      Xemb_bf, E_, Wih_bf, E_, Xall, G3_, nullptr, E_);

  // 5) cooperative GRU + attention scan (fills Hbf)
  scan_kernel<<<NBLK, 384, 0, stream>>>(Xall, gu, gp, gbb, bhh, cu, ci, Wa2, ba2,
                                        Whh_bf, Wa1h_bf, Hbf, part_su, part_si, cnt);

  // 6) logits = H @ Wv.T + bv  (M=4096 N=32000 K=512) -> d_out
  gemm_bf16_kernel<<<dim3(V_ / BNT, (B_ * T_) / BMT), 256, 0, stream>>>(
      Hbf, H_, Wv_bf, H_, out, V_, bv, H_);
}
